// SparseConv2d_46196668236075
// MI455X (gfx1250) — matrix operand order repaired and checked by reading, not compiled, not run on hardware
//
#include <hip/hip_runtime.h>

typedef __attribute__((ext_vector_type(16))) _Float16 v16h;
typedef __attribute__((ext_vector_type(8)))  _Float16 v8h;
typedef __attribute__((ext_vector_type(4)))  unsigned v4u;
typedef __attribute__((ext_vector_type(8)))  float    v8f;

#define IN_C   256
#define OUT_C  256
#define HW     56
#define IMG    (HW * HW)        // 3136 = 49*64 -> 64-row blocks never straddle images
#define KTOT   (IN_C * 9)       // 2304
#define BATCH  32
#define M_TOTAL (BATCH * IMG)   // 100352 = 1568 * 64
#define KSTEP  32
#define MTILE  64
#define AROW   264              // 256 + 8: 16B-aligned rows, minimal (2-way) bank load

#define WT_BYTES ((size_t)OUT_C * KTOT * 2)       // 1,179,648 (multiple of 256)

// ---------------------------------------------------------------------------
// Pre-pass 1: OIHW f32 weights -> f16 Wt[oc][k'] with k' = r*256 + ci (r=kh*3+kw).
// ---------------------------------------------------------------------------
__global__ __launch_bounds__(256) void weight_permute_f16(
    const float* __restrict__ w, _Float16* __restrict__ wt)
{
    const int idx = blockIdx.x * 256 + threadIdx.x;   // 589824 elements
    const int oc  = idx / KTOT;
    const int kp  = idx - oc * KTOT;
    const int r   = kp >> 8;
    const int ci  = kp & 255;
    wt[idx] = (_Float16)w[(size_t)oc * KTOT + ci * 9 + r];
}

// ---------------------------------------------------------------------------
// Pre-pass 2: x NCHW f32 -> NHWC f16:  xh[(n*3136 + hw)*256 + c]
// ---------------------------------------------------------------------------
__global__ __launch_bounds__(256) void x_nhwc_f16(
    const float* __restrict__ x, _Float16* __restrict__ xh)
{
    const size_t idx = (size_t)blockIdx.x * 256 + threadIdx.x;  // < 25,690,112
    const int c  = (int)(idx & 255);
    const int q  = (int)(idx >> 8);       // n*3136 + hw  (< 100352)
    const int n  = q / IMG;
    const int hw = q - n * IMG;
    xh[idx] = (_Float16)x[((size_t)n * IN_C + c) * IMG + hw];
}

// ---------------------------------------------------------------------------
// Implicit-GEMM conv2d 3x3 pad1: D[M=100352][N=256] = A[M][K'] * B[K'][N]
// Block: 256 thr = 8 waves. Block tile 64M x 256N, wave tile 64M x 32N
// (4 A frags x 2 B frags = 8 WMMA per K-chunk). Per tap r: stage 64 pixels x
// 256 ch from NHWC f16, then 8 K-chunks -> 64 WMMAs per barrier pair.
// ---------------------------------------------------------------------------
__global__ __launch_bounds__(256) void SparseConv2d_wmma_kernel(
    const _Float16* __restrict__ xh,    // [32][56][56][256] f16
    const _Float16* __restrict__ wt,    // [256][2304] permuted f16
    const float* __restrict__ bias,     // [256]
    float* __restrict__ out)            // [32][256][56][56]
{
    __shared__ _Float16 As[MTILE][AROW];   // 33.8 KB

    const int tid      = threadIdx.x;
    const int lane     = tid & 31;
    const int wave     = tid >> 5;
    const int half_sel = lane >> 4;
    const int mrow     = lane & 15;

    const int m_base   = blockIdx.x * MTILE;
    const int n_img    = m_base / IMG;            // uniform across block
    const int rem_base = m_base - n_img * IMG;

    // ---- staging assignment: pixel = tid&63, 64-channel slab = tid>>6 ----
    const int pix     = tid & 63;
    const int colbase = (tid >> 6) * 64;          // 0,64,128,192
    const int rp    = rem_base + pix;             // < 3136 always
    const int oh_st = rp / HW;
    const int ow_st = rp - oh_st * HW;
    const _Float16* ximg = xh + (size_t)n_img * IMG * IN_C;

    // ---- output columns for this wave ----
    const int oc0 = wave * 32 + mrow;
    const int oc1 = oc0 + 16;
    const _Float16* wrow0 = wt + (size_t)oc0 * KTOT + half_sel * 16;
    const _Float16* wrow1 = wt + (size_t)oc1 * KTOT + half_sel * 16;

    v8f acc[4][2];
    #pragma unroll
    for (int s = 0; s < 4; ++s) { acc[s][0] = (v8f){}; acc[s][1] = (v8f){}; }

    #pragma unroll 1
    for (int r = 0; r < 9; ++r) {                 // tap loop: im2col decode lives here
        const int kh = r / 3;
        const int kw = r - kh * 3;
        const int ih = oh_st + kh - 1;
        const int iw = ow_st + kw - 1;
        const bool inb = ((unsigned)ih < (unsigned)HW) & ((unsigned)iw < (unsigned)HW);

        // Branchless padding: safe base pointer + AND-mask zeroing.
        const _Float16* src =
            (inb ? ximg + (size_t)(ih * HW + iw) * IN_C : ximg) + colbase;
        const unsigned msk = inb ? 0xFFFFFFFFu : 0u;

        // ---- stage 64 channels: clause of 8x global_load_b128 -> 8x ds_store_b128
        #pragma unroll
        for (int g = 0; g < 8; ++g) {
            v4u pk = *(const v4u*)(src + g * 8);
            #pragma unroll
            for (int q = 0; q < 4; ++q) pk[q] &= msk;
            *(v4u*)(&As[pix][colbase + g * 8]) = pk;
        }
        __syncthreads();

        if (r < 8)
            __builtin_prefetch(wrow0 + (r + 1) * 256, 0, 1);

        // ---- 8 K-chunks, 8 WMMA each: 64 WMMAs per barrier pair ----
        #pragma unroll 2
        for (int cb = 0; cb < 8; ++cb) {
            const int kk0 = r * 256 + cb * KSTEP;

            const v16h bf0 = *(const v16h*)(wrow0 + kk0);   // 2x global_load_b128
            const v16h bf1 = *(const v16h*)(wrow1 + kk0);

            // Load ALL four A fragments first (distinct regs -> one ds clause,
            // one wait), then issue the 8 WMMAs back-to-back.
            v16h af[4];
            #pragma unroll
            for (int s = 0; s < 4; ++s) {
                const _Float16* rs = &As[mrow + s * 16][cb * KSTEP];
                const v8h lo = *(const v8h*)(rs + half_sel * 16);
                const v8h hi = *(const v8h*)(rs + half_sel * 16 + 8);
                #pragma unroll
                for (int q = 0; q < 8; ++q) { af[s][q] = lo[q]; af[s][q + 8] = hi[q]; }
            }

            #pragma unroll
            for (int s = 0; s < 4; ++s) {
                acc[s][0] = __builtin_amdgcn_wmma_f32_16x16x32_f16(
                    false, af[s], false, bf0, (short)0, acc[s][0], false, false);
                acc[s][1] = __builtin_amdgcn_wmma_f32_16x16x32_f16(
                    false, af[s], false, bf1, (short)0, acc[s][1], false, false);
            }
        }
        __syncthreads();
    }

    // ---- epilogue: bias + f32 NCHW store (no divisions: block-uniform image) ----
    const float bv0 = bias[oc0];
    const float bv1 = bias[oc1];
    float* ob0 = out + ((size_t)n_img * OUT_C + oc0) * IMG + rem_base;
    float* ob1 = out + ((size_t)n_img * OUT_C + oc1) * IMG + rem_base;
    #pragma unroll
    for (int s = 0; s < 4; ++s) {
        #pragma unroll
        for (int j = 0; j < 8; ++j) {
            const int l = s * 16 + j + 8 * half_sel;
            ob0[l] = acc[s][0][j] + bv0;
            ob1[l] = acc[s][1][j] + bv1;
        }
    }
}

extern "C" void kernel_launch(void* const* d_in, const int* in_sizes, int n_in,
                              void* d_out, int out_size, void* d_ws, size_t ws_size,
                              hipStream_t stream) {
    (void)in_sizes; (void)n_in; (void)ws_size; (void)out_size;
    const float* x    = (const float*)d_in[0];   // 32*256*56*56
    const float* w    = (const float*)d_in[1];   // 256*256*3*3
    const float* bias = (const float*)d_in[2];   // 256
    float* out        = (float*)d_out;

    _Float16* wt = (_Float16*)d_ws;                            // 1.18 MB
    _Float16* xh = (_Float16*)((char*)d_ws + WT_BYTES);        // 51.4 MB

    weight_permute_f16<<<dim3((OUT_C * KTOT) / 256), dim3(256), 0, stream>>>(w, wt);

    x_nhwc_f16<<<dim3((BATCH * IMG * IN_C) / 256), dim3(256), 0, stream>>>(x, xh);

    SparseConv2d_wmma_kernel<<<dim3(M_TOTAL / MTILE), dim3(256), 0, stream>>>(xh, wt, bias, out);
}
